// ResBlock_23235773072057
// MI455X (gfx1250) — matrix operand order repaired and checked by reading, not compiled, not run on hardware
//
#include <hip/hip_runtime.h>

// ---------------------------------------------------------------------------
// MI455X (gfx1250) sparse-conv SE residual block.
//  - conv as implicit GEMM on v_wmma_f32_16x16x32_bf16 (wave32)
//  - activations + packed weights are L2-resident; gathers hit L2, not HBM
//  - per-offset weight fragments staged into LDS via async global->LDS copies
//    (ASYNCcnt), double buffered; A fragments + neighbor indices software-
//    pipelined one k ahead; B fragments double-buffered in registers.
// ---------------------------------------------------------------------------

typedef __attribute__((ext_vector_type(16))) __bf16        v16bf;
typedef __attribute__((ext_vector_type(8)))  float         v8f;
typedef __attribute__((ext_vector_type(4)))  unsigned int  u32x4;

#define CH          96
#define KOFF        81
#define NBATCH      8
#define HIDDEN      24
#define FRAG_ELEMS  512                    // 32 lanes * 16 bf16 per fragment
#define FRAGS_PER_K 18                     // 3 c-chunks * 6 d-tiles
#define WK_ELEMS    (FRAGS_PER_K * FRAG_ELEMS)   // 9216 bf16 per offset
#define WK_VEC      (WK_ELEMS / 8)               // 1152 x b128 per offset
#define WFRAG_TOTAL (KOFF * WK_ELEMS)            // 746496 bf16 per conv
#define BN_EPS      1e-5f

union FragU { u32x4 q[2]; v16bf v; };

static __device__ __forceinline__ unsigned short f32_to_bf16(float f) {
  unsigned int u = __builtin_bit_cast(unsigned int, f);
  u += 0x7fffu + ((u >> 16) & 1u);        // round-to-nearest-even
  return (unsigned short)(u >> 16);
}

// ---------------- small elementwise / setup kernels ------------------------

__global__ void zero_small_kernel(float* s1, float* s2, float* pool) {
  int t = threadIdx.x;
  if (t < 2 * CH)      s1[t]  = 0.0f;
  if (t < 2 * CH)      s2[t]  = 0.0f;
  if (t < NBATCH * CH) pool[t] = 0.0f;
}

__global__ void pack_x_kernel(const float* __restrict__ src,
                              unsigned short* __restrict__ dst, int total) {
  int t = blockIdx.x * blockDim.x + threadIdx.x;
  if (t < total) dst[t] = f32_to_bf16(src[t]);
}

// Repack W[k][c][d] (f32) into per-offset WMMA B fragments (bf16).
// Fragment f = k*18 + cc*6 + dt holds the 32x16 B-tile (c-chunk cc, d-tile dt).
// Element (lane, i):  c = cc*32 + (lane>=16 ? 8 : 0) + (i&7) + (i>>3)*16, d = dt*16 + (lane&15).
__global__ void pack_w_kernel(const float* __restrict__ W1,
                              const float* __restrict__ W2,
                              unsigned short* __restrict__ F1,
                              unsigned short* __restrict__ F2) {
  int t = blockIdx.x * blockDim.x + threadIdx.x;
  if (t >= 2 * WFRAG_TOTAL) return;
  const float* W = (t < WFRAG_TOTAL) ? W1 : W2;
  unsigned short* F = (t < WFRAG_TOTAL) ? F1 : F2;
  int e    = (t < WFRAG_TOTAL) ? t : (t - WFRAG_TOTAL);
  int f    = e >> 9;
  int r    = e & 511;
  int lane = r >> 4;
  int i    = r & 15;
  int k    = f / FRAGS_PER_K;
  int rem  = f % FRAGS_PER_K;
  int cc   = rem / 6;
  int dt   = rem % 6;
  int hi   = lane >> 4;
  int c    = cc * 32 + hi * 8 + (i & 7) + (i >> 3) * 16;
  int d    = dt * 16 + (lane & 15);
  F[e] = f32_to_bf16(W[((size_t)k * CH + c) * CH + d]);
}

// ---------------- implicit-GEMM helpers ------------------------------------

// Async global->LDS staging of one k-offset's fragment block (18KB).
// dsaddr = low 32 bits of the flat shared pointer (LDS offset).
static __device__ __forceinline__ void stage_async(const unsigned short* WF,
                                                   int k, unsigned lbase,
                                                   int tid) {
  const char* gsrc = (const char*)(WF + (size_t)k * WK_ELEMS);
  for (int i = tid; i < WK_VEC; i += 256) {
    unsigned laddr = lbase + (unsigned)i * 16u;
    const char* gp = gsrc + (size_t)i * 16;
    asm volatile("global_load_async_to_lds_b128 %0, %1, off"
                 :: "v"(laddr), "v"(gp)
                 : "memory");
  }
}

static __device__ __forceinline__ void load_a(FragU a[3],
                                              const unsigned short* __restrict__ X,
                                              int idx, int hi) {
  const unsigned short* xr = X + (size_t)idx * CH;
#pragma unroll
  for (int cc = 0; cc < 3; ++cc) {
    const int c0 = cc * 32 + hi * 8;          // 16B-aligned
    a[cc].q[0] = *reinterpret_cast<const u32x4*>(xr + c0);
    a[cc].q[1] = *reinterpret_cast<const u32x4*>(xr + c0 + 16);
  }
}

static __device__ __forceinline__ void load_b(FragU& b,
                                              const unsigned short* wb,
                                              int frag, int lane) {
  const u32x4* bp = reinterpret_cast<const u32x4*>(
      wb + ((size_t)frag * FRAG_ELEMS + (size_t)lane * 16));
  b.q[0] = bp[0];
  b.q[1] = bp[1];
}

// ---------------- implicit-GEMM sparse conv (bf16 WMMA) --------------------
// Block: 256 threads = 8 waves. Wave w computes rows [r0, r0+16) x all 96 cols.
__global__ __launch_bounds__(256)
void conv_gemm_kernel(const unsigned short* __restrict__ X,
                      const unsigned short* __restrict__ WF,
                      const int* __restrict__ nbr,
                      float* __restrict__ Y,
                      float* __restrict__ stats,     // [0..95]=sum, [96..191]=sumsq
                      int npts) {
  __shared__ __align__(16) unsigned short sw[2][WK_ELEMS];
  __shared__ float s_sum[CH];
  __shared__ float s_sq[CH];

  const int tid  = threadIdx.x;
  const int wave = tid >> 5;
  const int lane = tid & 31;
  const int hi   = lane >> 4;          // which half-wave
  const int ln   = lane & 15;
  const int r0   = blockIdx.x * 128 + wave * 16;

  const unsigned lb0 = (unsigned)(size_t)&sw[0][0];
  const unsigned lb1 = (unsigned)(size_t)&sw[1][0];

  // stage W[0] fragments (ASYNCcnt path, no VGPR round-trip)
  stage_async(WF, 0, lb0, tid);

  const int rowA = r0 + ln;                       // A rows (shared by half-waves)
  const int rowV = (rowA < npts) ? rowA : 0;      // clamp OOB gathers
  const size_t nbase = (size_t)rowV * KOFF;

  // neighbor-index pipeline: idx_cur for k, idx_nxt for k+1
  int idx_cur = nbr[nbase + 0];
  int idx_nxt = nbr[nbase + 1];
  (void)idx_cur;

  FragU a_c[3], a_n[3];
  load_a(a_c, X, idx_cur, hi);

  v8f acc[6] = {};

  for (int k = 0; k < KOFF; ++k) {
    // our own async staging for sw[k&1] must have landed, then WG-wide barrier
    asm volatile("s_wait_asynccnt 0x0" ::: "memory");
    __syncthreads();

    if (k + 1 < KOFF) {
      stage_async(WF, k + 1, (k & 1) ? lb0 : lb1, tid);  // stage next buffer
      load_a(a_n, X, idx_nxt, hi);                       // prefetch next A
    }
    if (k + 2 < KOFF) idx_nxt = nbr[nbase + k + 2];      // prefetch next index

    const unsigned short* wb = &sw[k & 1][0];

    // B fragments double-buffered in registers: one fragment in flight ahead
    FragU b0, b1;
    load_b(b0, wb, 0, lane);
#pragma unroll
    for (int t = 0; t < FRAGS_PER_K; ++t) {
      if (t + 1 < FRAGS_PER_K) load_b((t & 1) ? b0 : b1, wb, t + 1, lane);
      const FragU& bc = (t & 1) ? b1 : b0;
      const int cc = t / 6;                 // A fragment reused 6x
      const int dt = t % 6;
      acc[dt] = __builtin_amdgcn_wmma_f32_16x16x32_bf16(
          false, a_c[cc].v, false, bc.v, (short)0, acc[dt], false, false);
    }

    if (k + 1 < KOFF) {
#pragma unroll
      for (int cc = 0; cc < 3; ++cc) a_c[cc] = a_n[cc];
    }
  }

  // ---- store + BN partial stats ----
  if (tid < CH) { s_sum[tid] = 0.0f; s_sq[tid] = 0.0f; }
  __syncthreads();

#pragma unroll
  for (int dt = 0; dt < 6; ++dt) {
    const int col = dt * 16 + ln;
    float ls = 0.0f, lq = 0.0f;
#pragma unroll
    for (int j = 0; j < 8; ++j) {
      const int row = r0 + hi * 8 + j;            // C/D layout: hi half holds M=8..15
      const float v = acc[dt][j];
      if (row < npts) {
        Y[(size_t)row * CH + col] = v;
        ls += v;
        lq += v * v;
      }
    }
    atomicAdd(&s_sum[col], ls);                   // ds_add_f32
    atomicAdd(&s_sq[col], lq);
  }
  __syncthreads();
  if (tid < CH) {
    atomicAdd(&stats[tid],      s_sum[tid]);
    atomicAdd(&stats[CH + tid], s_sq[tid]);
  }
}

// ---------------- BN (+ReLU) -> bf16 repack for conv2 ----------------------
__global__ void bn_relu_pack_kernel(const float* __restrict__ Y,
                                    const float* __restrict__ stats,
                                    const float* __restrict__ gamma,
                                    const float* __restrict__ beta,
                                    unsigned short* __restrict__ Xout,
                                    int total, float invN) {
  int t = blockIdx.x * blockDim.x + threadIdx.x;
  if (t >= total) return;
  int c = t % CH;
  float mean = stats[c] * invN;
  float var  = stats[CH + c] * invN - mean * mean;
  float xh   = gamma[c] * (Y[t] - mean) * rsqrtf(var + BN_EPS) + beta[c];
  xh = fmaxf(xh, 0.0f);
  Xout[t] = f32_to_bf16(xh);
}

// ---------------- BN2 (in place) + SE segment-sum pooling ------------------
__global__ void bn_pool_kernel(float* __restrict__ Y,
                               const float* __restrict__ stats,
                               const float* __restrict__ gamma,
                               const float* __restrict__ beta,
                               const int* __restrict__ bid,
                               float* __restrict__ pool,
                               int total, float invN) {
  __shared__ float lp[NBATCH * CH];
  for (int i = threadIdx.x; i < NBATCH * CH; i += 256) lp[i] = 0.0f;
  __syncthreads();

  int base = blockIdx.x * 8192;
  int end  = base + 8192; if (end > total) end = total;
  for (int e = base + threadIdx.x; e < end; e += 256) {
    int n = e / CH, c = e % CH;
    float mean = stats[c] * invN;
    float var  = stats[CH + c] * invN - mean * mean;
    float xh   = gamma[c] * (Y[e] - mean) * rsqrtf(var + BN_EPS) + beta[c];
    Y[e] = xh;
    atomicAdd(&lp[bid[n] * CH + c], xh);
  }
  __syncthreads();
  for (int i = threadIdx.x; i < NBATCH * CH; i += 256)
    if (lp[i] != 0.0f) atomicAdd(&pool[i], lp[i]);
}

// ---------------- SE FCs (single workgroup; tiny) --------------------------
__global__ void se_fc_kernel(const float* __restrict__ pool,
                             const int* __restrict__ bid,
                             const float* __restrict__ w1, const float* __restrict__ b1,
                             const float* __restrict__ w2, const float* __restrict__ b2,
                             float* __restrict__ sew, int npts) {
  __shared__ float cnt[NBATCH];
  __shared__ float pm[NBATCH * CH];
  __shared__ float hid[NBATCH * HIDDEN];

  if (threadIdx.x < NBATCH) cnt[threadIdx.x] = 0.0f;
  __syncthreads();
  for (int nn = threadIdx.x; nn < npts; nn += 256) atomicAdd(&cnt[bid[nn]], 1.0f);
  __syncthreads();

  for (int i = threadIdx.x; i < NBATCH * CH; i += 256)
    pm[i] = pool[i] / fmaxf(cnt[i / CH], 1.0f);
  __syncthreads();

  for (int i = threadIdx.x; i < NBATCH * HIDDEN; i += 256) {
    int b_ = i / HIDDEN, h = i % HIDDEN;
    float s = b1[h];
    for (int c = 0; c < CH; ++c) s += pm[b_ * CH + c] * w1[c * HIDDEN + h];
    hid[i] = fmaxf(s, 0.0f);
  }
  __syncthreads();

  for (int i = threadIdx.x; i < NBATCH * CH; i += 256) {
    int b_ = i / CH, d = i % CH;
    float s = b2[d];
    for (int h = 0; h < HIDDEN; ++h) s += hid[b_ * HIDDEN + h] * w2[h * CH + d];
    sew[i] = 1.0f / (1.0f + expf(-s));
  }
}

// ---------------- SE scale + residual + ReLU -------------------------------
__global__ void se_apply_kernel(const float* __restrict__ Y,
                                const float* __restrict__ sew,
                                const int* __restrict__ bid,
                                const float* __restrict__ feats,
                                float* __restrict__ out, int total) {
  int t = blockIdx.x * blockDim.x + threadIdx.x;
  if (t >= total) return;
  int n = t / CH, c = t % CH;
  out[t] = fmaxf(fmaf(Y[t], sew[bid[n] * CH + c], feats[t]), 0.0f);
}

// ---------------------------------------------------------------------------

extern "C" void kernel_launch(void* const* d_in, const int* in_sizes, int n_in,
                              void* d_out, int out_size, void* d_ws, size_t ws_size,
                              hipStream_t stream) {
  (void)n_in; (void)out_size; (void)ws_size;

  const float* feats = (const float*)d_in[0];
  const float* W1    = (const float*)d_in[1];
  const float* g1    = (const float*)d_in[2];
  const float* b1    = (const float*)d_in[3];
  const float* W2    = (const float*)d_in[4];
  const float* g2    = (const float*)d_in[5];
  const float* b2    = (const float*)d_in[6];
  const float* fc1w  = (const float*)d_in[7];
  const float* fc1b  = (const float*)d_in[8];
  const float* fc2w  = (const float*)d_in[9];
  const float* fc2b  = (const float*)d_in[10];
  const int*   nbr   = (const int*)d_in[11];
  const int*   bid   = (const int*)d_in[12];
  const int    n     = in_sizes[12];          // number of points
  const int    totalX = n * CH;

  // workspace carve-out (L2-resident working set; Y and X reused across convs)
  char* base = (char*)d_ws;
  size_t off = 0;
  auto carve = [&](size_t bytes) -> char* {
    char* p = base + off;
    off = (off + bytes + 255) & ~(size_t)255;
    return p;
  };
  unsigned short* X  = (unsigned short*)carve((size_t)totalX * 2);       // bf16 acts
  unsigned short* F1 = (unsigned short*)carve((size_t)WFRAG_TOTAL * 2);  // packed W1
  unsigned short* F2 = (unsigned short*)carve((size_t)WFRAG_TOTAL * 2);  // packed W2
  float* Y      = (float*)carve((size_t)totalX * 4);                     // f32 conv out
  float* stats1 = (float*)carve(2 * CH * 4);
  float* stats2 = (float*)carve(2 * CH * 4);
  float* pool   = (float*)carve(NBATCH * CH * 4);
  float* sew    = (float*)carve(NBATCH * CH * 4);

  const float invN = 1.0f / (float)n;
  const int gemmGrid = (n + 127) / 128;

  zero_small_kernel<<<1, 768, 0, stream>>>(stats1, stats2, pool);
  pack_x_kernel<<<(totalX + 255) / 256, 256, 0, stream>>>(feats, X, totalX);
  pack_w_kernel<<<(2 * WFRAG_TOTAL + 255) / 256, 256, 0, stream>>>(W1, W2, F1, F2);

  conv_gemm_kernel<<<gemmGrid, 256, 0, stream>>>(X, F1, nbr, Y, stats1, n);
  bn_relu_pack_kernel<<<(totalX + 255) / 256, 256, 0, stream>>>(Y, stats1, g1, b1, X, totalX, invN);
  conv_gemm_kernel<<<gemmGrid, 256, 0, stream>>>(X, F2, nbr, Y, stats2, n);

  bn_pool_kernel<<<(totalX + 8191) / 8192, 256, 0, stream>>>(Y, stats2, g2, b2, bid, pool, totalX, invN);
  se_fc_kernel<<<1, 256, 0, stream>>>(pool, bid, fc1w, fc1b, fc2w, fc2b, sew, n);
  se_apply_kernel<<<(totalX + 255) / 256, 256, 0, stream>>>(Y, sew, bid, feats, (float*)d_out, totalX);
}
